// TtAttention_82274393522664
// MI455X (gfx1250) — matrix operand order repaired and checked by reading, not compiled, not run on hardware
//
#include <hip/hip_runtime.h>
#include <hip/hip_bf16.h>

typedef __attribute__((ext_vector_type(16))) __bf16       v16bf;
typedef __attribute__((ext_vector_type(8)))  float        v8f;
typedef __attribute__((ext_vector_type(4)))  unsigned int u32x4;
typedef int v4i_vs __attribute__((vector_size(16)));   // matches builtin param type

#define B_   4
#define S_   1024
#define D_   4096
#define NH_  32
#define NKV_ 8
#define HD_  128

#if __has_builtin(__builtin_amdgcn_global_load_async_to_lds_b128) && \
    __has_builtin(__builtin_amdgcn_s_wait_asynccnt)
#define HAS_ASYNC_LDS 1
#else
#define HAS_ASYNC_LDS 0
#endif

union AB {
    v16bf v;
    unsigned int u[8];
};

__device__ __forceinline__ v8f wmma_bf16(const AB& a, const AB& b, v8f c) {
    return __builtin_amdgcn_wmma_f32_16x16x32_bf16(
        /*neg_a=*/false, a.v, /*neg_b=*/false, b.v,
        /*c_mod=*/(short)0, c, /*reuse_a=*/false, /*reuse_b=*/false);
}

// 16B global -> LDS copy. Async (ASYNCcnt-tracked, no VGPR roundtrip) when the
// gfx1250 builtin is available, else synchronous fallback.
// Generic LDS pointer low 32 bits == LDS byte offset (ISA aperture rule), so
// the integer-cast to addrspace(3) is valid; global generic ptr -> as1 is a
// value-preserving 64-bit cast.
__device__ __forceinline__ void copy_b128_to_lds(void* lds_dst, const void* gsrc) {
#if HAS_ASYNC_LDS
    __builtin_amdgcn_global_load_async_to_lds_b128(
        (__attribute__((address_space(1))) v4i_vs*)(unsigned long long)gsrc,
        (__attribute__((address_space(3))) v4i_vs*)(unsigned int)(unsigned long long)lds_dst,
        0, 0);
#else
    *(u32x4*)lds_dst = *(const u32x4*)gsrc;
#endif
}

__device__ __forceinline__ void wait_async_le4() {
#if HAS_ASYNC_LDS
    __builtin_amdgcn_s_wait_asynccnt(4);
#endif
}
__device__ __forceinline__ void wait_async_le0() {
#if HAS_ASYNC_LDS
    __builtin_amdgcn_s_wait_asynccnt(0);
#endif
}

// ---------------------------------------------------------------------------
// f32 -> bf16 conversion (grid-stride)
// ---------------------------------------------------------------------------
__global__ void cvt_kernel(const float* __restrict__ in, __bf16* __restrict__ out, long n) {
    long i = (long)blockIdx.x * blockDim.x + threadIdx.x;
    long stride = (long)gridDim.x * blockDim.x;
    for (; i < n; i += stride) out[i] = (__bf16)in[i];
}

// ---------------------------------------------------------------------------
// RoPE: f32 [B*S, H, HD] -> bf16 same layout; pairs (2i, 2i+1)
// ---------------------------------------------------------------------------
__global__ void rope_kernel(const float* __restrict__ t,
                            const float* __restrict__ fc,
                            const float* __restrict__ fs,
                            __bf16* __restrict__ out, int H, long npairs) {
    long i = (long)blockIdx.x * blockDim.x + threadIdx.x;
    if (i >= npairs) return;
    int hd2 = (int)(i % (HD_ / 2));
    long row = i / (HD_ / 2);          // ((b*S + s)*H + h)
    long r2 = row / H;
    int s = (int)(r2 % S_);
    long base = row * HD_;
    float tr = t[base + 2 * hd2];
    float ti = t[base + 2 * hd2 + 1];
    float c = fc[(long)s * (HD_ / 2) + hd2];
    float sn = fs[(long)s * (HD_ / 2) + hd2];
    out[base + 2 * hd2]     = (__bf16)(tr * c - ti * sn);
    out[base + 2 * hd2 + 1] = (__bf16)(tr * sn + ti * c);
}

// ---------------------------------------------------------------------------
// C[M,N](f32) = A[M,K](bf16) @ W[N,K]^T(bf16)
// Block tile 128x128, 8 waves (wave32), wave tile 32x64 = 2x4 WMMA 16x16 tiles.
// Double-buffered LDS, async global->LDS staging (4 async ops / wave / slab):
// issue slab t+1, s_wait_asynccnt 4 retires slab t while t+1 stays in flight.
// ---------------------------------------------------------------------------
__global__ __launch_bounds__(256) void gemm_bf16_kernel(
    const __bf16* __restrict__ A, const __bf16* __restrict__ W,
    float* __restrict__ C, int M, int N, int K) {
    __shared__ unsigned int lAs[2][128][20];   // 128 rows x 32 bf16 (16 uints) + pad
    __shared__ unsigned int lBs[2][128][20];

    const int tid  = threadIdx.x;
    const int lane = tid & 31;
    const int wid  = tid >> 5;
    const int half = lane >> 4;
    const int l16  = lane & 15;
    const int wm   = wid & 3;     // 4 waves along M
    const int wn   = wid >> 2;    // 2 waves along N
    const int bm   = blockIdx.y * 128;
    const int bn   = blockIdx.x * 128;

    // this thread's two (row, chunk) slots for tile staging
    const int row0 = tid >> 2,          c80 = tid & 3;
    const int row1 = (tid + 256) >> 2,  c81 = (tid + 256) & 3;

    v8f acc[2][4];
    for (int mi = 0; mi < 2; ++mi)
        for (int ni = 0; ni < 4; ++ni)
            for (int r = 0; r < 8; ++r) acc[mi][ni][r] = 0.0f;

    const int T = K / 32;

    // prologue: stage slab 0 into buffer 0
    copy_b128_to_lds(&lAs[0][row0][c80 * 4], A + (long)(bm + row0) * K + c80 * 8);
    copy_b128_to_lds(&lBs[0][row0][c80 * 4], W + (long)(bn + row0) * K + c80 * 8);
    copy_b128_to_lds(&lAs[0][row1][c81 * 4], A + (long)(bm + row1) * K + c81 * 8);
    copy_b128_to_lds(&lBs[0][row1][c81 * 4], W + (long)(bn + row1) * K + c81 * 8);

    for (int t = 0; t < T; ++t) {
        const int kt = t * 32;
        if (t + 1 < T) {
            const int kn = kt + 32, nb = (t + 1) & 1;
            copy_b128_to_lds(&lAs[nb][row0][c80 * 4], A + (long)(bm + row0) * K + kn + c80 * 8);
            copy_b128_to_lds(&lBs[nb][row0][c80 * 4], W + (long)(bn + row0) * K + kn + c80 * 8);
            copy_b128_to_lds(&lAs[nb][row1][c81 * 4], A + (long)(bm + row1) * K + kn + c81 * 8);
            copy_b128_to_lds(&lBs[nb][row1][c81 * 4], W + (long)(bn + row1) * K + kn + c81 * 8);
            wait_async_le4();   // slab t complete; slab t+1 still streaming
            __builtin_prefetch(A + (long)(bm + row0) * K + kn + 32, 0, 1);
        } else {
            wait_async_le0();
        }
        __syncthreads();        // slab t visible to all waves

        const int cb = t & 1;
        AB afr[2];
        for (int mi = 0; mi < 2; ++mi) {
            int m = wm * 32 + mi * 16 + l16;
            for (int v = 0; v < 8; ++v) {
                int k2 = ((v < 4) ? v : v + 4) + half * 4;  // A 16-bit layout
                afr[mi].u[v] = lAs[cb][m][k2];
            }
        }
        AB bfr[4];
        for (int ni = 0; ni < 4; ++ni) {
            int n = wn * 64 + ni * 16 + l16;
            for (int j = 0; j < 8; ++j)
                bfr[ni].u[j] = lBs[cb][n][((j < 4) ? j : j + 4) + half * 4];    // B 16-bit layout
        }
        for (int mi = 0; mi < 2; ++mi)
            for (int ni = 0; ni < 4; ++ni)
                acc[mi][ni] = wmma_bf16(afr[mi], bfr[ni], acc[mi][ni]);
        __syncthreads();        // reads done before buffer is overwritten (t+2)
    }

    for (int mi = 0; mi < 2; ++mi)
        for (int ni = 0; ni < 4; ++ni)
            for (int r = 0; r < 8; ++r) {
                int row = bm + wm * 32 + mi * 16 + half * 8 + r;
                int col = bn + wn * 64 + ni * 16 + l16;
                C[(long)row * N + col] = acc[mi][ni][r];
            }
}

// ---------------------------------------------------------------------------
// Flash attention: one block per (b, h, 128-row q tile). 8 waves; each wave
// owns 16 q rows. Keys streamed in 64-key tiles (K async to LDS, V transposed
// manually). Causal mask (sliding window 4096 >= S).
// ---------------------------------------------------------------------------
__global__ __launch_bounds__(256) void attn_kernel(
    const __bf16* __restrict__ qb, const __bf16* __restrict__ kb,
    const __bf16* __restrict__ vb, __bf16* __restrict__ ob) {
    __shared__ unsigned int sK[64][68];      // [key][hd] 64x128 bf16 + pad
    __shared__ unsigned int sVT[128][36];    // [hd][key] transposed 128x64 bf16 + pad
    __shared__ unsigned int sP[8][16][34];   // per-wave P 16x64 bf16 + pad

    const int tid  = threadIdx.x;
    const int lane = tid & 31;
    const int w    = tid >> 5;
    const int half = lane >> 4;
    const int l16  = lane & 15;

    int blk = blockIdx.x;
    const int qt = blk & 7;  blk >>= 3;
    const int h  = blk & 31; const int b = blk >> 5;
    const int hkv = h >> 2;                   // NH/NKV = 4
    const float scale = 0.08838834764831845f; // 1/sqrt(128)

    // Q fragments resident in registers (A-matrix 16x32 layout), 4 hd chunks
    AB qf[4];
    {
        int s = qt * 128 + w * 16 + l16;
        const unsigned int* qu =
            (const unsigned int*)qb + ((long)(b * S_ + s) * NH_ + h) * (HD_ / 2);
        for (int kk = 0; kk < 4; ++kk)
            for (int v = 0; v < 8; ++v) {
                int k2 = ((v < 4) ? v : v + 4) + half * 4;
                qf[kk].u[v] = qu[kk * 16 + k2];
            }
    }

    v8f o_acc[8];
    for (int ht = 0; ht < 8; ++ht)
        for (int r = 0; r < 8; ++r) o_acc[ht][r] = 0.0f;
    float m_r[8], l_r[8];
    for (int r = 0; r < 8; ++r) { m_r[r] = -1e30f; l_r[r] = 0.0f; }

    const int nkt = 2 * qt + 2;  // key tiles of 64 covering keys <= qt*128+127
    for (int kt = 0; kt < nkt; ++kt) {
        __syncthreads();   // previous iteration's LDS reads complete
        // load 64x128 K tile (async) + transposed V tile (manual scatter)
        for (int i = 0; i < 4; ++i) {
            int idx = tid + i * 256;       // 1024 chunks of 8 bf16
            int row = idx >> 4;            // key in tile
            int c8  = idx & 15;            // hd chunk
            long kbase = ((long)(b * S_ + kt * 64 + row) * NKV_ + hkv) * HD_;
            copy_b128_to_lds(&sK[row][c8 * 4], kb + kbase + c8 * 8);
            u32x4 vv = *(const u32x4*)(vb + kbase + c8 * 8);
            const __bf16* vh = (const __bf16*)&vv;
            for (int j = 0; j < 8; ++j)
                ((__bf16*)sVT)[(c8 * 8 + j) * 72 + row] = vh[j];
        }
        wait_async_le0();
        __syncthreads();

        // scores: 16 q rows x 64 keys = 4 n-tiles, K-dim = HD (4 chunks)
        v8f s_acc[4];
        for (int nt = 0; nt < 4; ++nt) {
            for (int r = 0; r < 8; ++r) s_acc[nt][r] = 0.0f;
            int key = nt * 16 + l16;
            for (int kk = 0; kk < 4; ++kk) {
                AB bf;
                for (int j = 0; j < 8; ++j)
                    bf.u[j] = sK[key][kk * 16 + ((j < 4) ? j : j + 4) + half * 4];
                s_acc[nt] = wmma_bf16(qf[kk], bf, s_acc[nt]);
            }
        }

        // scale + causal mask + online softmax (rows live across 16-lane halves)
        float pm[8];
        for (int r = 0; r < 8; ++r) pm[r] = -1e30f;
        const int qi_base = qt * 128 + w * 16 + half * 8;
        for (int nt = 0; nt < 4; ++nt) {
            int ki = kt * 64 + nt * 16 + l16;
            for (int r = 0; r < 8; ++r) {
                float val = s_acc[nt][r] * scale;
                if (ki > qi_base + r) val = -1e9f;
                s_acc[nt][r] = val;
                pm[r] = fmaxf(pm[r], val);
            }
        }
        for (int r = 0; r < 8; ++r)
            for (int off = 1; off < 16; off <<= 1)
                pm[r] = fmaxf(pm[r], __shfl_xor(pm[r], off, 32));
        float alpha[8];
        for (int r = 0; r < 8; ++r) {
            float mn = fmaxf(m_r[r], pm[r]);
            alpha[r] = __expf(m_r[r] - mn);
            m_r[r] = mn;
        }
        float psum[8];
        for (int r = 0; r < 8; ++r) psum[r] = 0.0f;
        __bf16* sPh = (__bf16*)sP[w];
        for (int nt = 0; nt < 4; ++nt) {
            int key = nt * 16 + l16;
            for (int r = 0; r < 8; ++r) {
                float p = __expf(s_acc[nt][r] - m_r[r]);
                psum[r] += p;
                sPh[(half * 8 + r) * 68 + key] = (__bf16)p;
            }
        }
        for (int r = 0; r < 8; ++r)
            for (int off = 1; off < 16; off <<= 1)
                psum[r] += __shfl_xor(psum[r], off, 32);
        for (int r = 0; r < 8; ++r) l_r[r] = l_r[r] * alpha[r] + psum[r];
        for (int ht = 0; ht < 8; ++ht)
            for (int r = 0; r < 8; ++r) o_acc[ht][r] *= alpha[r];
        __syncthreads();

        // O += P(16x64) @ V(64x128): 2 key chunks of 32, 8 hd tiles
        for (int kc = 0; kc < 2; ++kc) {
            AB pf;
            for (int v = 0; v < 8; ++v) {
                int k2 = ((v < 4) ? v : v + 4) + half * 4;
                pf.u[v] = sP[w][l16][kc * 16 + k2];
            }
            for (int ht = 0; ht < 8; ++ht) {
                AB vf;
                int hd = ht * 16 + l16;
                for (int j = 0; j < 8; ++j)
                    vf.u[j] = sVT[hd][kc * 16 + ((j < 4) ? j : j + 4) + half * 4];
                o_acc[ht] = wmma_bf16(pf, vf, o_acc[ht]);
            }
        }
    }

    // epilogue: normalize and store bf16 [b, s, h, hd]
    int s0 = qt * 128 + w * 16 + half * 8;
    for (int ht = 0; ht < 8; ++ht)
        for (int r = 0; r < 8; ++r) {
            float o = o_acc[ht][r] / l_r[r];
            ob[((long)(b * S_ + s0 + r) * NH_ + h) * HD_ + ht * 16 + l16] = (__bf16)o;
        }
}

// ---------------------------------------------------------------------------
// Launch: x->bf16; Q/K/V GEMMs (+RoPE); flash attention; output projection.
// ---------------------------------------------------------------------------
extern "C" void kernel_launch(void* const* d_in, const int* in_sizes, int n_in,
                              void* d_out, int out_size, void* d_ws, size_t ws_size,
                              hipStream_t stream) {
    const float* x  = (const float*)d_in[0];
    const float* wq = (const float*)d_in[1];
    const float* wk = (const float*)d_in[2];
    const float* wv = (const float*)d_in[3];
    const float* wo = (const float*)d_in[4];
    const float* fc = (const float*)d_in[5];
    const float* fs = (const float*)d_in[6];
    // d_in[7] mask (pure causal here), d_in[8] positions, d_in[9] seqlen: unused

    char* ws = (char*)d_ws;
    const size_t MB = 1024ull * 1024ull;
    __bf16* xb   = (__bf16*)(ws + 0);          // 32 MB  x in bf16
    __bf16* wbuf = (__bf16*)(ws + 32  * MB);   // 32 MB  current weight in bf16
    float*  regA = (float*) (ws + 64  * MB);   // 64 MB  f32 GEMM output
    __bf16* qb   = (__bf16*)(ws + 128 * MB);   // 32 MB  q (roped) bf16
    __bf16* kb   = (__bf16*)(ws + 160 * MB);   //  8 MB  k (roped) bf16
    __bf16* vb   = (__bf16*)(ws + 168 * MB);   //  8 MB  v bf16
    __bf16* aob  = (__bf16*)(ws + 176 * MB);   // 32 MB  attention output bf16

    const long nx = (long)B_ * S_ * D_;  // 16.7M
    cvt_kernel<<<4096, 256, 0, stream>>>(x, xb, nx);

    // Q = x @ wq^T  -> RoPE -> qb
    cvt_kernel<<<4096, 256, 0, stream>>>(wq, wbuf, (long)NH_ * HD_ * D_);
    gemm_bf16_kernel<<<dim3(32, 32), 256, 0, stream>>>(xb, wbuf, regA, 4096, 4096, 4096);
    long qpairs = (long)B_ * S_ * NH_ * (HD_ / 2);
    rope_kernel<<<(int)((qpairs + 255) / 256), 256, 0, stream>>>(regA, fc, fs, qb, NH_, qpairs);

    // K = x @ wk^T  -> RoPE -> kb
    cvt_kernel<<<1024, 256, 0, stream>>>(wk, wbuf, (long)NKV_ * HD_ * D_);
    gemm_bf16_kernel<<<dim3(8, 32), 256, 0, stream>>>(xb, wbuf, regA, 4096, 1024, 4096);
    long kpairs = (long)B_ * S_ * NKV_ * (HD_ / 2);
    rope_kernel<<<(int)((kpairs + 255) / 256), 256, 0, stream>>>(regA, fc, fs, kb, NKV_, kpairs);

    // V = x @ wv^T -> vb
    cvt_kernel<<<1024, 256, 0, stream>>>(wv, wbuf, (long)NKV_ * HD_ * D_);
    gemm_bf16_kernel<<<dim3(8, 32), 256, 0, stream>>>(xb, wbuf, regA, 4096, 1024, 4096);
    cvt_kernel<<<1024, 256, 0, stream>>>(regA, vb, (long)B_ * S_ * NKV_ * HD_);

    // attention
    attn_kernel<<<B_ * NH_ * 8, 256, 0, stream>>>(qb, kb, vb, aob);

    // out = attn_out @ wo^T  (f32 to d_out)
    cvt_kernel<<<4096, 256, 0, stream>>>(wo, wbuf, (long)D_ * NH_ * HD_);
    gemm_bf16_kernel<<<dim3(32, 32), 256, 0, stream>>>(aob, wbuf, (float*)d_out, 4096, 4096, 4096);
}
